// EmbeddingSharedWeights_74509092651513
// MI455X (gfx1250) — matrix operand order repaired and checked by reading, not compiled, not run on hardware
//
#include <hip/hip_runtime.h>
#include <hip/hip_bf16.h>

#define HIDDEN 1024
#define TWOH   2048
#define BM 128
#define BN 128
#define BK 32
#define LDK 40   // padded LDS row stride in bf16 elements (80B, 16B-aligned)

typedef __attribute__((ext_vector_type(16))) __bf16 v16bf;
typedef __attribute__((ext_vector_type(8)))  float  v8f;

union Frag32 { uint4 q[2]; v16bf v; };

// round-to-nearest-even fp32 -> bf16 bits
__device__ __forceinline__ unsigned bf16_rne_bits(float f) {
  unsigned u = __float_as_uint(f);
  return (u + 0x7FFFu + ((u >> 16) & 1u)) >> 16;
}

// Split two fp32 into packed (hi,hi) and (lo,lo) bf16 pairs.
// hi = truncated bf16 (exact in fp32), lo = RNE bf16 of the exact residual.
__device__ __forceinline__ void split2(float x0, float x1, unsigned& hi, unsigned& lo) {
  unsigned u0 = __float_as_uint(x0) & 0xFFFF0000u;
  unsigned u1 = __float_as_uint(x1) & 0xFFFF0000u;
  float r0 = x0 - __uint_as_float(u0);
  float r1 = x1 - __uint_as_float(u1);
  hi = (u0 >> 16) | (u1 & 0xFFFF0000u);
  lo = bf16_rne_bits(r0) | (bf16_rne_bits(r1) << 16);
}

__global__ __launch_bounds__(256)
void EmbeddingSharedWeights_74509092651513_kernel(const int* __restrict__ inputs,
                                                  const float* __restrict__ W0,
                                                  const float* __restrict__ W1,
                                                  float* __restrict__ out) {
  __shared__ __align__(16) unsigned short Ahi[BM][LDK];
  __shared__ __align__(16) unsigned short Alo[BM][LDK];
  __shared__ __align__(16) unsigned short Bhi[BN][LDK];  // stored transposed: [n][k]
  __shared__ __align__(16) unsigned short Blo[BN][LDK];
  __shared__ int tok[BM];

  const int tid  = threadIdx.x;
  const int lane = tid & 31;
  const int wave = tid >> 5;

  // wave -> 32x64 subtile of the 128x128 block tile
  const int mW = (wave & 3) * 32;
  const int nW = (wave >> 2) * 64;
  const int lane15   = lane & 15;
  const int laneHalf = lane >> 4;

  // stage token ids for this M-tile
  if (tid < BM) tok[tid] = inputs[blockIdx.y * BM + tid];
  __syncthreads();

  v8f acc[2][4] = {};

  // A staging: 2 threads per row, each covers 16 contiguous k (4x float4)
  const int aRow  = tid >> 1;
  const int aHalf = tid & 1;
  const float* aSrcBase = W0 + (size_t)tok[aRow] * TWOH + aHalf * 16;

  for (int k0 = 0; k0 < TWOH; k0 += BK) {
    // ---- stage A tile (gathered rows of W0), split fp32 -> bf16 hi/lo ----
    {
      const float* src = aSrcBase + k0;
      if (k0 + BK < TWOH) __builtin_prefetch(src + BK, 0, 0);  // global_prefetch_b8
      #pragma unroll
      for (int i = 0; i < 4; ++i) {
        float4 f = *(const float4*)(src + i * 4);
        unsigned h0, l0, h1, l1;
        split2(f.x, f.y, h0, l0);
        split2(f.z, f.w, h1, l1);
        const int e = aHalf * 16 + i * 4;  // element offset, byte off multiple of 8
        unsigned* ph = (unsigned*)&Ahi[aRow][e];
        unsigned* pl = (unsigned*)&Alo[aRow][e];
        ph[0] = h0; ph[1] = h1;
        pl[0] = l0; pl[1] = l1;
      }
    }
    // ---- stage B tile (W1 rows), transpose into [n][k] hi/lo ----
    #pragma unroll
    for (int i = 0; i < 4; ++i) {
      const int c    = tid + i * 256;   // 1024 float4 chunks total
      const int kr   = c >> 5;          // 0..31
      const int col4 = (c & 31) * 4;    // 0..124
      float4 f = *(const float4*)(W1 + (size_t)(k0 + kr) * HIDDEN
                                     + blockIdx.x * BN + col4);
      unsigned h0, l0, h1, l1;
      split2(f.x, f.y, h0, l0);
      split2(f.z, f.w, h1, l1);
      Bhi[col4 + 0][kr] = (unsigned short)(h0 & 0xFFFF);
      Bhi[col4 + 1][kr] = (unsigned short)(h0 >> 16);
      Bhi[col4 + 2][kr] = (unsigned short)(h1 & 0xFFFF);
      Bhi[col4 + 3][kr] = (unsigned short)(h1 >> 16);
      Blo[col4 + 0][kr] = (unsigned short)(l0 & 0xFFFF);
      Blo[col4 + 1][kr] = (unsigned short)(l0 >> 16);
      Blo[col4 + 2][kr] = (unsigned short)(l1 & 0xFFFF);
      Blo[col4 + 3][kr] = (unsigned short)(l1 >> 16);
    }
    __syncthreads();

    // ---- load fragments from LDS (layouts per ISA 7.12.2) ----
    // A 16x32 bf16: lanes 0-15 -> M=lane, K = 0-7 & 16-23; lanes 16-31 -> K = 8-15 & 24-31
    Frag32 Ah[2], Al[2];
    const int aK = laneHalf * 8;
    #pragma unroll
    for (int mi = 0; mi < 2; ++mi) {
      const unsigned short* ph = &Ahi[mW + mi * 16 + lane15][aK];
      const unsigned short* pl = &Alo[mW + mi * 16 + lane15][aK];
      Ah[mi].q[0] = *(const uint4*)(ph);
      Ah[mi].q[1] = *(const uint4*)(ph + 16);
      Al[mi].q[0] = *(const uint4*)(pl);
      Al[mi].q[1] = *(const uint4*)(pl + 16);
    }
    // B 32x16 bf16: lane holds column N=lane&15, K = 0-7 & 16-23 (lanes 0-15) or 8-15 & 24-31 (lanes 16-31)
    Frag32 Bh[4], Bl[4];
    const int bK = laneHalf * 8;
    #pragma unroll
    for (int ni = 0; ni < 4; ++ni) {
      const unsigned short* ph = &Bhi[nW + ni * 16 + lane15][bK];
      const unsigned short* pl = &Blo[nW + ni * 16 + lane15][bK];
      Bh[ni].q[0] = *(const uint4*)(ph);
      Bh[ni].q[1] = *(const uint4*)(ph + 16);
      Bl[ni].q[0] = *(const uint4*)(pl);
      Bl[ni].q[1] = *(const uint4*)(pl + 16);
    }

    // ---- 24 WMMAs: bf16x3 split-precision accumulation ----
    #pragma unroll
    for (int mi = 0; mi < 2; ++mi) {
      #pragma unroll
      for (int ni = 0; ni < 4; ++ni) {
        acc[mi][ni] = __builtin_amdgcn_wmma_f32_16x16x32_bf16(
            false, Ah[mi].v, false, Bh[ni].v, (short)0, acc[mi][ni], false, false);
        acc[mi][ni] = __builtin_amdgcn_wmma_f32_16x16x32_bf16(
            false, Ah[mi].v, false, Bl[ni].v, (short)0, acc[mi][ni], false, false);
        acc[mi][ni] = __builtin_amdgcn_wmma_f32_16x16x32_bf16(
            false, Al[mi].v, false, Bh[ni].v, (short)0, acc[mi][ni], false, false);
      }
    }
    __syncthreads();  // protect LDS before next stage overwrites
  }

  // ---- epilogue: mask padding token, scale by sqrt(1024)=32 exactly ----
  #pragma unroll
  for (int mi = 0; mi < 2; ++mi) {
    #pragma unroll
    for (int ni = 0; ni < 4; ++ni) {
      const int nGlob = blockIdx.x * BN + nW + ni * 16 + lane15;
      #pragma unroll
      for (int r = 0; r < 8; ++r) {
        const int rLoc = mW + mi * 16 + r + laneHalf * 8;  // C/D layout: row = r + 8*(lane>=16)
        const float s = (tok[rLoc] != 0) ? 32.0f : 0.0f;
        out[(size_t)(blockIdx.y * BM + rLoc) * HIDDEN + nGlob] = acc[mi][ni][r] * s;
      }
    }
  }
}

extern "C" void kernel_launch(void* const* d_in, const int* in_sizes, int n_in,
                              void* d_out, int out_size, void* d_ws, size_t ws_size,
                              hipStream_t stream) {
  const int*   inputs = (const int*)d_in[0];
  const float* W0     = (const float*)d_in[1];
  const float* W1     = (const float*)d_in[2];
  float*       out    = (float*)d_out;

  const int M = in_sizes[0];           // 4 * 4096 = 16384 tokens
  dim3 grid(HIDDEN / BN, M / BM);      // (8, 128)
  EmbeddingSharedWeights_74509092651513_kernel<<<grid, 256, 0, stream>>>(inputs, W0, W1, out);
}
